// _IPEXGatedMLPMOEXPU_55834574848354
// MI455X (gfx1250) — compile-verified, matrix operand order checked by reading, not run on hardware
//
#include <hip/hip_runtime.h>
#include <hip/hip_bf16.h>

// Problem constants (match reference)
#define TT 4096
#define EE 8
#define HH 2048
#define II 4096

typedef __attribute__((ext_vector_type(16))) __bf16          v16bf;
typedef __attribute__((ext_vector_type(2)))  __bf16          vbf2;
typedef __attribute__((ext_vector_type(8)))  float           v8f;
typedef __attribute__((ext_vector_type(2)))  float           vf2;
typedef __attribute__((ext_vector_type(8)))  unsigned short  ushort8;

struct U16x16 { ushort8 lo, hi; };

static __device__ __forceinline__ v16bf combine(ushort8 lo, ushort8 hi) {
    U16x16 t{lo, hi};
    return __builtin_bit_cast(v16bf, t);
}

// fp32 -> bf16 (hardware cvt, RNE)
static __device__ __forceinline__ unsigned short f2bf(float f) {
    __bf16 h = (__bf16)f;
    return __builtin_bit_cast(unsigned short, h);
}

// two fp32 -> packed 2x bf16 (v_cvt_pk_bf16_f32 when available)
static __device__ __forceinline__ unsigned int pack2(float a, float b) {
    vf2 f = {a, b};
    vbf2 v = __builtin_convertvector(f, vbf2);
    return __builtin_bit_cast(unsigned int, v);
}

static __device__ __forceinline__ v8f wmma_bf16(v16bf a, v16bf b, v8f c) {
    // (neg_a, A, neg_b, B, c_mod, C, reuse_a, reuse_b)
    return __builtin_amdgcn_wmma_f32_16x16x32_bf16(false, a, false, b, (short)0, c, false, false);
}

// ---------------------------------------------------------------- zero utils
__global__ void zero_f32_kernel(float* __restrict__ p, int n4) {
    int i = blockIdx.x * blockDim.x + threadIdx.x;
    if (i < n4) {
        float4 z = {0.f, 0.f, 0.f, 0.f};
        ((float4*)p)[i] = z;
    }
}

__global__ void zero_counts_kernel(int* __restrict__ counts) {
    if (threadIdx.x < EE) counts[threadIdx.x] = 0;
}

// ---------------------------------------------------------------- routing
__global__ void router_kernel(const float* __restrict__ logits,
                              int* __restrict__ counts,
                              int* __restrict__ tok_ids,
                              float* __restrict__ tok_w) {
    int t = blockIdx.x * blockDim.x + threadIdx.x;
    if (t >= TT) return;
    float l[EE];
    float m = -3.0e38f;
    #pragma unroll
    for (int e = 0; e < EE; ++e) { l[e] = logits[t * EE + e]; m = fmaxf(m, l[e]); }
    int i1 = 0; float b1 = -3.0e38f;
    #pragma unroll
    for (int e = 0; e < EE; ++e) if (l[e] > b1) { b1 = l[e]; i1 = e; }
    int i2 = 0; float b2 = -3.0e38f;
    #pragma unroll
    for (int e = 0; e < EE; ++e) if (e != i1 && l[e] > b2) { b2 = l[e]; i2 = e; }
    float p1 = __expf(b1 - m), p2 = __expf(b2 - m);
    float s = p1 + p2;
    int s1 = atomicAdd(&counts[i1], 1);
    tok_ids[i1 * TT + s1] = t; tok_w[i1 * TT + s1] = p1 / s;
    int s2 = atomicAdd(&counts[i2], 1);
    tok_ids[i2 * TT + s2] = t; tok_w[i2 * TT + s2] = p2 / s;
}

// ---------------------------------------------------------------- x: fp32->bf16
__global__ void cvt_kernel(const float* __restrict__ x, unsigned short* __restrict__ xb) {
    int i = (blockIdx.x * blockDim.x + threadIdx.x) * 4;
    float4 v = *(const float4*)(x + i);
    uint2 q;
    q.x = pack2(v.x, v.y);
    q.y = pack2(v.z, v.w);
    *(uint2*)(xb + i) = q;
}

// ---------------------------------------------------------------- GEMM1 + SiLU·up
// Block: 128 gathered token rows x 64 act cols (=128 gemm cols: gate+up).
// Wave w: row tiles {2(w&3), 2(w&3)+1}, act col tiles {2(w>>2), 2(w>>2)+1}.
// 8 accumulators/wave -> 8 WMMAs per K-step (K step = 32).
__global__ __launch_bounds__(256) void gemm1_kernel(
    const unsigned short* __restrict__ Xb,   // [T,H] bf16 bits
    const float* __restrict__ W13e,          // [H, 2I] fp32
    const int* __restrict__ countp,
    const int* __restrict__ toks,            // [T]
    unsigned short* __restrict__ act)        // [T, I] bf16 bits (slot-indexed)
{
    const int cnt = *countp;
    const int row0 = blockIdx.y * 128;
    if (row0 >= cnt) return;
    const int j0 = blockIdx.x * 64;

    __shared__ __align__(16) unsigned short ldsB[128][40];  // [col][k], padded 80B rows
    __shared__ int ldsTok[128];

    const int tid = threadIdx.x;
    if (tid < 128) {
        int s = row0 + tid;
        ldsTok[tid] = (s < cnt) ? toks[s] : toks[0];  // clamp: finite pad rows
    }
    __syncthreads();

    const int wave = tid >> 5, lane = tid & 31;
    const int rbase = (wave & 3) * 32;        // two row tiles: rbase, rbase+16
    const int cg = wave >> 2;                 // act col tiles 2cg, 2cg+1
    const int ln16 = lane & 15;
    const int khalf = (lane < 16) ? 0 : 8;    // K-half (A/B) == M offset (C/D)

    const unsigned short* arow0 = Xb + (size_t)ldsTok[rbase + ln16] * HH;
    const unsigned short* arow1 = Xb + (size_t)ldsTok[rbase + 16 + ln16] * HH;

    // staging coords: 4x4 block per thread: cols c4..c4+3 at k4..k4+3
    const int c4 = (tid & 31) * 4;
    const int k4 = (tid >> 5) * 4;
    const int colg = (c4 < 64) ? (j0 + c4) : (II + j0 + (c4 - 64));

    v8f G[2][2] = {{{}, {}}, {{}, {}}};
    v8f U[2][2] = {{{}, {}}, {{}, {}}};

    for (int h0 = 0; h0 < HH; h0 += 32) {
        // ---- stage 32x128 fp32 tile -> bf16 col-major LDS (vectorized both ends)
        const float* src = W13e + (size_t)(h0 + k4) * (2 * II) + colg;
        float4 r0 = *(const float4*)(src);
        float4 r1 = *(const float4*)(src + 2 * II);
        float4 r2 = *(const float4*)(src + 4 * II);
        float4 r3 = *(const float4*)(src + 6 * II);
        uint2 q;
        q.x = pack2(r0.x, r1.x); q.y = pack2(r2.x, r3.x); *(uint2*)&ldsB[c4 + 0][k4] = q;
        q.x = pack2(r0.y, r1.y); q.y = pack2(r2.y, r3.y); *(uint2*)&ldsB[c4 + 1][k4] = q;
        q.x = pack2(r0.z, r1.z); q.y = pack2(r2.z, r3.z); *(uint2*)&ldsB[c4 + 2][k4] = q;
        q.x = pack2(r0.w, r1.w); q.y = pack2(r2.w, r3.w); *(uint2*)&ldsB[c4 + 3][k4] = q;
        if (h0 + 32 < HH)
            __builtin_prefetch(src + (size_t)32 * (2 * II), 0, 0);
        __syncthreads();

        v16bf a0 = combine(*(const ushort8*)(arow0 + h0 + khalf),
                           *(const ushort8*)(arow0 + h0 + khalf + 16));
        v16bf a1 = combine(*(const ushort8*)(arow1 + h0 + khalf),
                           *(const ushort8*)(arow1 + h0 + khalf + 16));
        #pragma unroll
        for (int j = 0; j < 2; ++j) {
            const int ct = cg * 2 + j;
            v16bf bg = combine(*(const ushort8*)&ldsB[ct * 16 + ln16][khalf],
                               *(const ushort8*)&ldsB[ct * 16 + ln16][khalf + 16]);
            v16bf bu = combine(*(const ushort8*)&ldsB[64 + ct * 16 + ln16][khalf],
                               *(const ushort8*)&ldsB[64 + ct * 16 + ln16][khalf + 16]);
            G[0][j] = wmma_bf16(a0, bg, G[0][j]);
            G[1][j] = wmma_bf16(a1, bg, G[1][j]);
            U[0][j] = wmma_bf16(a0, bu, U[0][j]);
            U[1][j] = wmma_bf16(a1, bu, U[1][j]);
        }
        __syncthreads();
    }

    // fused SiLU(gate)*up -> bf16 act (zeros for pad slots)
    #pragma unroll
    for (int i = 0; i < 2; ++i) {
        const int mbase = rbase + i * 16 + khalf;
        #pragma unroll
        for (int r = 0; r < 8; ++r) {
            int slot = row0 + mbase + r;
            bool ok = slot < cnt;
            #pragma unroll
            for (int j = 0; j < 2; ++j) {
                float g = G[i][j][r], u = U[i][j][r];
                float v = ok ? (g / (1.f + __expf(-g))) * u : 0.f;
                act[(size_t)slot * II + j0 + (cg * 2 + j) * 16 + ln16] = f2bf(v);
            }
        }
    }
}

// ---------------------------------------------------------------- GEMM2 + scatter-add
// Block: 128 slots x 64 H cols. Wave w: row tiles {2(w&3),2(w&3)+1},
// col tiles {2(w>>2),2(w>>2)+1}; 4 accumulators -> 4 WMMAs per K-step.
__global__ __launch_bounds__(256) void gemm2_kernel(
    const unsigned short* __restrict__ act,  // [T, I] bf16 bits
    const float* __restrict__ W2e,           // [I, H] fp32
    const int* __restrict__ countp,
    const int* __restrict__ toks,
    const float* __restrict__ tws,
    float* __restrict__ out)                 // [T, H] fp32
{
    const int cnt = *countp;
    const int row0 = blockIdx.y * 128;
    if (row0 >= cnt) return;
    const int j0 = blockIdx.x * 64;

    __shared__ __align__(16) unsigned short ldsB[64][40];
    __shared__ int   ldsTok[128];
    __shared__ float ldsW[128];

    const int tid = threadIdx.x;
    if (tid < 128) {
        int s = row0 + tid;
        bool ok = s < cnt;
        ldsTok[tid] = ok ? toks[s] : 0;
        ldsW[tid]   = ok ? tws[s] : 0.f;
    }
    __syncthreads();

    const int wave = tid >> 5, lane = tid & 31;
    const int rbase = (wave & 3) * 32;
    const int cg = wave >> 2;
    const int ln16 = lane & 15;
    const int khalf = (lane < 16) ? 0 : 8;

    const unsigned short* arow0 = act + (size_t)(row0 + rbase + ln16) * II;
    const unsigned short* arow1 = act + (size_t)(row0 + rbase + 16 + ln16) * II;

    // staging coords: 4 cols x 2 k per thread
    const int c4 = (tid & 15) * 4;
    const int k2 = (tid >> 4) * 2;

    v8f acc[2][2] = {{{}, {}}, {{}, {}}};

    for (int k0 = 0; k0 < II; k0 += 32) {
        const float* src = W2e + (size_t)(k0 + k2) * HH + j0 + c4;
        float4 r0 = *(const float4*)(src);
        float4 r1 = *(const float4*)(src + HH);
        *(unsigned int*)&ldsB[c4 + 0][k2] = pack2(r0.x, r1.x);
        *(unsigned int*)&ldsB[c4 + 1][k2] = pack2(r0.y, r1.y);
        *(unsigned int*)&ldsB[c4 + 2][k2] = pack2(r0.z, r1.z);
        *(unsigned int*)&ldsB[c4 + 3][k2] = pack2(r0.w, r1.w);
        if (k0 + 32 < II)
            __builtin_prefetch(src + (size_t)32 * HH, 0, 0);
        __syncthreads();

        v16bf a0 = combine(*(const ushort8*)(arow0 + k0 + khalf),
                           *(const ushort8*)(arow0 + k0 + khalf + 16));
        v16bf a1 = combine(*(const ushort8*)(arow1 + k0 + khalf),
                           *(const ushort8*)(arow1 + k0 + khalf + 16));
        #pragma unroll
        for (int j = 0; j < 2; ++j) {
            const int ct = cg * 2 + j;
            v16bf b = combine(*(const ushort8*)&ldsB[ct * 16 + ln16][khalf],
                              *(const ushort8*)&ldsB[ct * 16 + ln16][khalf + 16]);
            acc[0][j] = wmma_bf16(a0, b, acc[0][j]);
            acc[1][j] = wmma_bf16(a1, b, acc[1][j]);
        }
        __syncthreads();
    }

    #pragma unroll
    for (int i = 0; i < 2; ++i) {
        const int mbase = rbase + i * 16 + khalf;
        #pragma unroll
        for (int r = 0; r < 8; ++r) {
            int sl = mbase + r;
            int slot = row0 + sl;
            if (slot < cnt) {
                int tok = ldsTok[sl];
                float w = ldsW[sl];
                size_t o = (size_t)tok * HH + j0 + cg * 32 + ln16;
                atomicAdd(&out[o],      w * acc[i][0][r]);
                atomicAdd(&out[o + 16], w * acc[i][1][r]);
            }
        }
    }
}

// ---------------------------------------------------------------- launch
extern "C" void kernel_launch(void* const* d_in, const int* in_sizes, int n_in,
                              void* d_out, int out_size, void* d_ws, size_t ws_size,
                              hipStream_t stream) {
    const float* x      = (const float*)d_in[0];  // [T,H]
    const float* logits = (const float*)d_in[1];  // [T,E]
    const float* W13    = (const float*)d_in[2];  // [E,H,2I]
    const float* W2     = (const float*)d_in[3];  // [E,I,H]
    float* out = (float*)d_out;

    // workspace layout
    char* ws = (char*)d_ws;
    int*            counts  = (int*)ws;                                   // E ints (pad 256B)
    int*            tok_ids = (int*)(ws + 256);                           // E*T ints
    float*          tok_w   = (float*)(ws + 256 + (size_t)EE * TT * 4);   // E*T floats
    unsigned short* Xb      = (unsigned short*)(ws + 256 + (size_t)EE * TT * 8);  // T*H bf16
    unsigned short* act     = Xb + (size_t)TT * HH;                               // T*I bf16

    zero_counts_kernel<<<1, 32, 0, stream>>>(counts);
    zero_f32_kernel<<<(TT * HH / 4 + 255) / 256, 256, 0, stream>>>(out, TT * HH / 4);
    router_kernel<<<TT / 256, 256, 0, stream>>>(logits, counts, tok_ids, tok_w);
    cvt_kernel<<<TT * HH / 4 / 256, 256, 0, stream>>>(x, Xb);

    for (int e = 0; e < EE; ++e) {
        gemm1_kernel<<<dim3(II / 64, TT / 128), 256, 0, stream>>>(
            Xb, W13 + (size_t)e * HH * 2 * II, counts + e, tok_ids + (size_t)e * TT, act);
        gemm2_kernel<<<dim3(HH / 64, TT / 128), 256, 0, stream>>>(
            act, W2 + (size_t)e * II * HH, counts + e, tok_ids + (size_t)e * TT,
            tok_w + (size_t)e * TT, out);
    }
}
